// OrthogonalBasisSelfAttention_17712445129300
// MI455X (gfx1250) — matrix operand order repaired and checked by reading, not compiled, not run on hardware
//
#include <hip/hip_runtime.h>
#include <hip/hip_bf16.h>
#include <math.h>
#include <stdint.h>

// ---------------- problem constants (fixed by the reference) ----------------
constexpr int B_  = 2;
constexpr int L_  = 2048;
constexpr int D_  = 1024;
constexpr int H_  = 16;
constexpr int DH_ = 64;
constexpr int R_  = 512;
constexpr int NB_ = 6;
constexpr float INV_SQRT_R = 0.04419417382415922f;  // 1/sqrt(512)
constexpr float PI_F = 3.14159265358979323846f;

// ---------------- vector types for WMMA ----------------
typedef __attribute__((ext_vector_type(16))) __bf16 v16bf;
typedef __attribute__((ext_vector_type(8)))  __bf16 v8bf;
typedef __attribute__((ext_vector_type(8)))  float  v8f;

#define DEV __device__ __forceinline__

// Build a 16-wide bf16 fragment from two 16-byte chunks (handles both the
// contiguous A-layout and B-layout halves (p, p+8)).
DEV v16bf ld16x2(const __bf16* p0, const __bf16* p1) {
  v8bf a = *(const v8bf*)p0;
  v8bf b = *(const v8bf*)p1;
  return __builtin_shufflevector(a, b, 0,1,2,3,4,5,6,7,8,9,10,11,12,13,14,15);
}

DEV v8f wmma_bf16(v16bf a, v16bf b, v8f c) {
  // D = A(16x32 bf16) x B(32x16 bf16) + C(16x16 f32)
  return __builtin_amdgcn_wmma_f32_16x16x32_bf16(
      /*neg_a=*/false, a, /*neg_b=*/false, b,
      /*c_mod=*/(short)0, c, /*reuse_a=*/false, /*reuse_b=*/false);
}

// Async copy 16 bytes global -> LDS (per lane). GV mode, tracked by ASYNCcnt.
DEV void async_g2l_b128(uint32_t lds_off, const void* gaddr) {
  asm volatile("global_load_async_to_lds_b128 %0, %1, off"
               :: "v"(lds_off), "v"(gaddr) : "memory");
}
DEV void wait_asynccnt0() { asm volatile("s_wait_asynccnt 0x0" ::: "memory"); }
DEV void wait_dscnt0()    { asm volatile("s_wait_dscnt 0x0" ::: "memory"); }

// ---------------- workspace layout ----------------
constexpr size_t alignup(size_t x) { return (x + 255) & ~(size_t)255; }
constexpr size_t OFF_HS = 0;                                                 // hs bf16 [B*L, D]
constexpr size_t OFF_WK = alignup(OFF_HS + (size_t)B_ * L_ * D_ * 2);        // Wk bf16 [D, D]
constexpr size_t OFF_WV = alignup(OFF_WK + (size_t)D_ * D_ * 2);             // Wv bf16
constexpr size_t OFF_WE = alignup(OFF_WV + (size_t)D_ * D_ * 2);             // WeffT bf16 [D(out), D(in)]
constexpr size_t OFF_M  = alignup(OFF_WE + (size_t)D_ * D_ * 2);             // M f32 [H, 64, 64]
constexpr size_t OFF_BQ = alignup(OFF_M  + (size_t)H_ * DH_ * DH_ * 4);      // b_eff f32 [D]
constexpr size_t OFF_BT = alignup(OFF_BQ + (size_t)D_ * 4);                  // bias table f32 [H, 4095]
constexpr size_t OFF_QM = alignup(OFF_BT + (size_t)H_ * 4095 * 4);           // qm bf16 [B,H,L,64]
constexpr size_t OFF_K  = alignup(OFF_QM + (size_t)B_ * H_ * L_ * DH_ * 2);  // k  bf16 [B,H,L,64]
constexpr size_t OFF_VT = alignup(OFF_K  + (size_t)B_ * H_ * L_ * DH_ * 2);  // vT bf16 [B,H,64,L]

// ---------------- prep kernels (tiny, scalar) ----------------
__global__ void k_cvt(const float* __restrict__ src, __bf16* __restrict__ dst, int n) {
  int i = blockIdx.x * blockDim.x + threadIdx.x;
  if (i < n) dst[i] = (__bf16)src[i];
}

// M[h][a][c] = sum_r Pq[h][a][r] * Pk[h][c][r]
__global__ void k_headmix(const float* __restrict__ Pq, const float* __restrict__ Pk,
                          float* __restrict__ M) {
  int t = blockIdx.x * blockDim.x + threadIdx.x;  // h*4096 + a*64 + c
  int h = t >> 12, a = (t >> 6) & 63, c = t & 63;
  const float* pa = Pq + (size_t)(h * DH_ + a) * R_;
  const float* pc = Pk + (size_t)(h * DH_ + c) * R_;
  float acc = 0.f;
  for (int r = 0; r < R_; ++r) acc = fmaf(pa[r], pc[r], acc);
  M[t] = acc;
}

// WeffT[o=h*64+c][d] = (1/sqrt(R)) * sum_a Wq[h*64+a][d] * M[h][a][c]
__global__ void k_weff(const float* __restrict__ Wq, const float* __restrict__ M,
                       __bf16* __restrict__ WeffT) {
  int t = blockIdx.x * blockDim.x + threadIdx.x;  // o*1024 + d
  int o = t >> 10, d = t & 1023;
  int h = o >> 6, c = o & 63;
  float acc = 0.f;
  for (int a = 0; a < DH_; ++a)
    acc = fmaf(Wq[(size_t)(h * DH_ + a) * D_ + d], M[h * 4096 + a * 64 + c], acc);
  WeffT[t] = (__bf16)(acc * INV_SQRT_R);
}

// b_eff[o=h*64+c] = (1/sqrt(R)) * sum_a bq[h*64+a] * M[h][a][c]
__global__ void k_bqeff(const float* __restrict__ bq, const float* __restrict__ M,
                        float* __restrict__ beff) {
  int o = blockIdx.x * blockDim.x + threadIdx.x;
  if (o >= D_) return;
  int h = o >> 6, c = o & 63;
  float acc = 0.f;
  for (int a = 0; a < DH_; ++a)
    acc = fmaf(bq[h * DH_ + a], M[h * 4096 + a * 64 + c], acc);
  beff[o] = acc * INV_SQRT_R;
}

// bias_table[h][idx] = sum_k coef[h][k] * cos(pi*k*(idx-2047)/2047),  idx = (j-i)+2047
__global__ void k_btable(const float* __restrict__ coef, float* __restrict__ bt) {
  int t = blockIdx.x * blockDim.x + threadIdx.x;
  if (t >= H_ * 4095) return;
  int h = t / 4095, idx = t % 4095;
  float x = (float)(idx - (L_ - 1)) * (1.0f / (float)(L_ - 1));
  float acc = 0.f;
  for (int kk = 0; kk < NB_; ++kk)
    acc = fmaf(coef[h * NB_ + kk], cosf(PI_F * (float)kk * x), acc);
  bt[t] = acc;
}

// ---------------- fused QKV projection (WMMA GEMM, ping-pong pipelined) ----------------
// mi=0: qm = hs @ WeffT^T + b_eff  -> [B,H,L,64] bf16   (scores projection folded in)
// mi=1: k  = hs @ Wk^T   + bk     -> [B,H,L,64] bf16
// mi=2: v  = hs @ Wv^T   + bv     -> [B,H,64,L] bf16    (transposed for P@V B-fragments)
__global__ __launch_bounds__(256) void k_qkv(
    const __bf16* __restrict__ hs,
    const __bf16* __restrict__ WeffT, const float* __restrict__ beff,
    const __bf16* __restrict__ Wk,    const float* __restrict__ bk,
    const __bf16* __restrict__ Wv,    const float* __restrict__ bv,
    __bf16* __restrict__ qm, __bf16* __restrict__ kout, __bf16* __restrict__ vT) {
  const int wid  = threadIdx.x >> 5;
  const int lane = threadIdx.x & 31;
  const int l15  = lane & 15;
  const int hi   = lane >> 4;
  const int gw   = blockIdx.x * 8 + wid;       // 0..12287
  const int mi   = gw / 4096;                  // which matrix
  const int tile = gw % 4096;
  const int rt   = tile >> 4;                  // row tile (16 rows), 0..255
  const int cg   = tile & 15;                  // col group (64 cols), 0..15

  const __bf16* Wt   = (mi == 0) ? WeffT : ((mi == 1) ? Wk : Wv);
  const float*  bias = (mi == 0) ? beff  : ((mi == 1) ? bk : bv);

  const int row = rt * 16 + l15;
  const __bf16* ap = hs + (size_t)row * D_;
  const __bf16* wp = Wt + (size_t)(cg * 64 + l15) * D_ + hi * 16;
  const int kb = hi * 16;                      // A-layout K contiguous per lane-half

  v8f acc[4];
#pragma unroll
  for (int n = 0; n < 4; ++n)
#pragma unroll
    for (int r = 0; r < 8; ++r) acc[n][r] = 0.f;

  auto loadFrags = [&](int kc, v16bf& a, v16bf* bfr) {
    const __bf16* a0 = ap + kc * 32;
    a = ld16x2(a0 + kb, a0 + kb + 8);
#pragma unroll
    for (int n = 0; n < 4; ++n) {
      const __bf16* bp = wp + (size_t)(n * 16) * D_ + kc * 32;
      bfr[n] = ld16x2(bp, bp + 8);
    }
  };

  // Ping-pong double buffering with 2x unroll: no register rotation copies;
  // each half consumes one buffer while the other buffer's loads are in
  // flight, so loadcnt waits sit a full stage behind the WMMAs.
  v16bf aA, bA[4], aB, bB[4];
  loadFrags(0, aA, bA);
  loadFrags(1, aB, bB);
  for (int kc = 0; kc < D_ / 32; kc += 2) {
#pragma unroll
    for (int n = 0; n < 4; ++n) acc[n] = wmma_bf16(aA, bA[n], acc[n]);
    if (kc + 2 < D_ / 32) loadFrags(kc + 2, aA, bA);
#pragma unroll
    for (int n = 0; n < 4; ++n) acc[n] = wmma_bf16(aB, bB[n], acc[n]);
    if (kc + 3 < D_ / 32) loadFrags(kc + 3, aB, bB);
  }

#pragma unroll
  for (int n = 0; n < 4; ++n) {
    const int col = cg * 64 + n * 16 + l15;
    const float bval = bias[col];
    const int h = col >> 6, d = col & 63;
#pragma unroll
    for (int r = 0; r < 8; ++r) {
      const int grow = rt * 16 + r + hi * 8;
      const int b = grow >> 11;         // / L_
      const int l = grow & (L_ - 1);
      const float val = acc[n][r] + bval;
      if (mi == 2)
        vT[((size_t)(b * H_ + h) * DH_ + d) * L_ + l] = (__bf16)val;
      else if (mi == 0)
        qm[((size_t)(b * H_ + h) * L_ + l) * DH_ + d] = (__bf16)val;
      else
        kout[((size_t)(b * H_ + h) * L_ + l) * DH_ + d] = (__bf16)val;
    }
  }
}

// ---------------- flash attention (WMMA + async LDS staging) ----------------
// Block = 4 waves, all sharing one (b,h); wave w owns 16 query rows. The K
// tile (32x64 bf16) and V tile (64x32 bf16) for each 32-column block are
// fetched ONCE per block with GLOBAL_LOAD_ASYNC_TO_LDS_B128, double-buffered
// one iteration ahead (ASYNCcnt + barrier), and consumed by all 4 waves as
// WMMA B-fragments straight from LDS.
__global__ __launch_bounds__(128) void k_attn(
    const __bf16* __restrict__ qm, const __bf16* __restrict__ kmat,
    const __bf16* __restrict__ vT, const float* __restrict__ btab,
    const float* __restrict__ mask, float* __restrict__ out) {
  __shared__ __bf16 kt[2][32 * 64];     // K tile: [col 0..31][d 0..63], 4 KB each
  __shared__ __bf16 vt[2][64 * 32];     // V tile: [d 0..63][col 0..31], 4 KB each
  __shared__ __bf16 pbuf[4][16 * 32];   // per-wave P staging, 1 KB each

  const int tid  = threadIdx.x;
  const int wid  = tid >> 5;
  const int lane = tid & 31;
  const int l15  = lane & 15;
  const int hi   = lane >> 4;
  const int gw   = blockIdx.x * 4 + wid;   // 0..4095
  const int bh   = gw >> 7;                // (b*H + h); identical for all 4 waves
  const int rt   = gw & 127;               // per-wave row tile within L
  const int b    = bh >> 4;
  const int h    = bh & 15;
  const size_t bhL = (size_t)bh * L_;

  const __bf16* kmat_bh = kmat + bhL * DH_;           // [L][64]
  const __bf16* vT_bh   = vT + (size_t)bh * DH_ * L_; // [64][L]

  // Block-cooperative async tile fetch: 512 x 16B chunks, 4 per thread.
  auto issue_tiles = [&](int jj, int buf) {
#pragma unroll
    for (int q = 0; q < 2; ++q) {        // K tile: chunk = row*8 + col16
      int ck = tid + q * 128;
      int c = ck >> 3, d0 = (ck & 7) * 8;
      uint32_t loff = (uint32_t)(uintptr_t)(&kt[buf][0]) + (uint32_t)ck * 16u;
      async_g2l_b128(loff, kmat_bh + (size_t)(jj + c) * DH_ + d0);
    }
#pragma unroll
    for (int q = 0; q < 2; ++q) {        // V tile: chunk = d*4 + col16
      int cv = tid + q * 128;
      int d = cv >> 2, c0 = (cv & 3) * 8;
      uint32_t loff = (uint32_t)(uintptr_t)(&vt[buf][0]) + (uint32_t)cv * 16u;
      async_g2l_b128(loff, vT_bh + (size_t)d * L_ + jj + c0);
    }
  };

  // qm A-fragments for K=64 (two K=32 chunks), resident for the whole loop
  const int row = rt * 16 + l15;
  const __bf16* qp = qm + (bhL + row) * DH_;
  const int kb = hi * 16;
  const v16bf aq0 = ld16x2(qp + kb,      qp + kb + 8);
  const v16bf aq1 = ld16x2(qp + 32 + kb, qp + 32 + kb + 8);

  float mrow[8], lrow[8];
  v8f O[4];
#pragma unroll
  for (int r = 0; r < 8; ++r) { mrow[r] = -3.0e38f; lrow[r] = 0.f; }
#pragma unroll
  for (int n = 0; n < 4; ++n)
#pragma unroll
    for (int r = 0; r < 8; ++r) O[n][r] = 0.f;

  const float* bth = btab + h * 4095;
  const float* mk  = mask + b * L_;
  __bf16* pb = pbuf[wid];

  issue_tiles(0, 0);                     // prime the pipeline

  for (int jj = 0; jj < L_; jj += 32) {
    const int cur = (jj >> 5) & 1;
    wait_asynccnt0();                    // own chunks landed in LDS
    __syncthreads();                     // everyone's chunks landed
    if (jj + 32 < L_) issue_tiles(jj + 32, cur ^ 1);

    const __bf16* kl = kt[cur];
    const __bf16* vl = vt[cur];

    // ---- issue ALL operand loads up front: K frags, V frags, bias, mask ----
    const __bf16* kp0 = kl + l15 * 64 + hi * 16;
    v16bf b00 = ld16x2(kp0,      kp0 + 8);
    v16bf b01 = ld16x2(kp0 + 32, kp0 + 40);
    const __bf16* kp1 = kl + (16 + l15) * 64 + hi * 16;
    v16bf b10 = ld16x2(kp1,      kp1 + 8);
    v16bf b11 = ld16x2(kp1 + 32, kp1 + 40);

    v16bf vfrag[4];
#pragma unroll
    for (int n = 0; n < 4; ++n) {
      const __bf16* vp = vl + (n * 16 + l15) * 32 + hi * 16;
      vfrag[n] = ld16x2(vp, vp + 8);
    }

    const int j0 = jj + l15, j1 = jj + 16 + l15;
    const float mv0 = mk[j0], mv1 = mk[j1];
    float bias0[8], bias1[8];
#pragma unroll
    for (int r = 0; r < 8; ++r) {
      const int i = rt * 16 + r + hi * 8;
      bias0[r] = bth[j0 - i + (L_ - 1)];
      bias1[r] = bth[j1 - i + (L_ - 1)];
    }

    // ---- S = qm @ k^T for two 16-column tiles ----
    v8f S0, S1;
#pragma unroll
    for (int r = 0; r < 8; ++r) { S0[r] = 0.f; S1[r] = 0.f; }
    S0 = wmma_bf16(aq0, b00, S0);
    S0 = wmma_bf16(aq1, b01, S0);
    S1 = wmma_bf16(aq0, b10, S1);
    S1 = wmma_bf16(aq1, b11, S1);

    // ---- bias + mask + online softmax (row stats across 16-lane halves) ----
#pragma unroll
    for (int r = 0; r < 8; ++r) {
      float s0 = S0[r] + bias0[r] + mv0;
      float s1 = S1[r] + bias1[r] + mv1;
      float rmax = fmaxf(s0, s1);
#pragma unroll
      for (int mm = 1; mm < 16; mm <<= 1) rmax = fmaxf(rmax, __shfl_xor(rmax, mm, 32));
      const float nm = fmaxf(mrow[r], rmax);
      const float al = __expf(mrow[r] - nm);
      const float p0 = __expf(s0 - nm);
      const float p1 = __expf(s1 - nm);
      float rs = p0 + p1;
#pragma unroll
      for (int mm = 1; mm < 16; mm <<= 1) rs += __shfl_xor(rs, mm, 32);
      lrow[r] = lrow[r] * al + rs;
      mrow[r] = nm;
#pragma unroll
      for (int n = 0; n < 4; ++n) O[n][r] *= al;
      const int prow = r + hi * 8;
      pb[prow * 32 + l15]      = (__bf16)p0;
      pb[prow * 32 + 16 + l15] = (__bf16)p1;
    }
    wait_dscnt0();                       // same-wave LDS RAW on pbuf

    // ---- reload P in A-layout, accumulate O += P @ V ----
    v16bf pA = ld16x2(&pb[l15 * 32 + kb], &pb[l15 * 32 + kb + 8]);
#pragma unroll
    for (int n = 0; n < 4; ++n) O[n] = wmma_bf16(pA, vfrag[n], O[n]);
  }

  // ---- epilogue: normalize and store ctx[b, l, h*64 + d] ----
#pragma unroll
  for (int r = 0; r < 8; ++r) {
    const float inv = 1.0f / lrow[r];
    const int i = rt * 16 + r + hi * 8;
    const size_t obase = ((size_t)b * L_ + i) * D_ + h * DH_;
#pragma unroll
    for (int n = 0; n < 4; ++n)
      out[obase + n * 16 + l15] = O[n][r] * inv;
  }
}

// ---------------- launcher ----------------
extern "C" void kernel_launch(void* const* d_in, const int* in_sizes, int n_in,
                              void* d_out, int out_size, void* d_ws, size_t ws_size,
                              hipStream_t stream) {
  (void)in_sizes; (void)n_in; (void)out_size; (void)ws_size;
  const float* hs = (const float*)d_in[0];
  const float* mk = (const float*)d_in[1];
  const float* Wq = (const float*)d_in[2];
  const float* bq = (const float*)d_in[3];
  const float* Wk = (const float*)d_in[4];
  const float* bk = (const float*)d_in[5];
  const float* Wv = (const float*)d_in[6];
  const float* bv = (const float*)d_in[7];
  const float* Pq = (const float*)d_in[8];
  const float* Pk = (const float*)d_in[9];
  const float* cf = (const float*)d_in[10];
  float* out = (float*)d_out;
  char*  w   = (char*)d_ws;

  __bf16* hsb  = (__bf16*)(w + OFF_HS);
  __bf16* wkb  = (__bf16*)(w + OFF_WK);
  __bf16* wvb  = (__bf16*)(w + OFF_WV);
  __bf16* weff = (__bf16*)(w + OFF_WE);
  float*  Mb   = (float*)(w + OFF_M);
  float*  bqe  = (float*)(w + OFF_BQ);
  float*  btab = (float*)(w + OFF_BT);
  __bf16* qmb  = (__bf16*)(w + OFF_QM);
  __bf16* kbuf = (__bf16*)(w + OFF_K);
  __bf16* vtb  = (__bf16*)(w + OFF_VT);

  const int nhs = B_ * L_ * D_;
  const int nw  = D_ * D_;
  k_cvt<<<(nhs + 255) / 256, 256, 0, stream>>>(hs, hsb, nhs);
  k_cvt<<<(nw + 255) / 256, 256, 0, stream>>>(Wk, wkb, nw);
  k_cvt<<<(nw + 255) / 256, 256, 0, stream>>>(Wv, wvb, nw);
  k_headmix<<<(H_ * DH_ * DH_) / 256, 256, 0, stream>>>(Pq, Pk, Mb);
  k_weff<<<nw / 256, 256, 0, stream>>>(Wq, Mb, weff);
  k_bqeff<<<(D_ + 255) / 256, 256, 0, stream>>>(bq, Mb, bqe);
  k_btable<<<(H_ * 4095 + 255) / 256, 256, 0, stream>>>(cf, btab);
  // 3 matrices * 256 row-tiles * 16 col-groups = 12288 waves, 8 waves/block
  k_qkv<<<1536, 256, 0, stream>>>(hsb, weff, bqe, wkb, bk, wvb, bv, qmb, kbuf, vtb);
  // B*H*(L/16) = 4096 waves, 4 waves/block
  k_attn<<<1024, 128, 0, stream>>>(qmb, kbuf, vtb, btab, mk, out);
}
